// TitanBlock_16080357556302
// MI455X (gfx1250) — matrix operand order repaired and checked by reading, not compiled, not run on hardware
//
#include <hip/hip_runtime.h>
#include <hip/hip_bf16.h>

typedef __attribute__((ext_vector_type(16))) _Float16 v16h;
typedef __attribute__((ext_vector_type(8)))  _Float16 v8h;
typedef __attribute__((ext_vector_type(8)))  float    v8f;
typedef int v4i __attribute__((vector_size(16)));

#define BB     2
#define LL     2048
#define DIMC   1024
#define NHEADS 16
#define HEADD  64
#define MTOK   (BB * LL)
#define LN_EPS 1e-5f

#if __has_builtin(__builtin_amdgcn_global_load_async_to_lds_b128) && \
    __has_builtin(__builtin_amdgcn_s_wait_asynccnt)
#define USE_ASYNC_LDS 1
#else
#define USE_ASYNC_LDS 0
#endif

__device__ __forceinline__ v16h cat16(v8h lo, v8h hi) {
  v16h r;
#pragma unroll
  for (int i = 0; i < 8; ++i) { r[i] = lo[i]; r[8 + i] = hi[i]; }
  return r;
}

__device__ __forceinline__ float sigm(float x) { return 1.0f / (1.0f + __expf(-x)); }

#if USE_ASYNC_LDS
__device__ __forceinline__ void async_cp_b128(const _Float16* g, _Float16* l) {
  __builtin_amdgcn_global_load_async_to_lds_b128(
      (__attribute__((address_space(1))) v4i*)g,
      (__attribute__((address_space(3))) v4i*)l, 0, 0);
}
__device__ __forceinline__ void wait_async0() {
  __builtin_amdgcn_s_wait_asynccnt(0);
}
#endif

// ---------------------------------------------------------------- fp32 -> f16
__global__ void cvt_f32_f16(const float* __restrict__ in, _Float16* __restrict__ out, int n) {
  int i = blockIdx.x * blockDim.x + threadIdx.x;
  if (i < n) out[i] = (_Float16)in[i];
}

// ------------------------------------------------- block-cooperative WMMA GEMM
// C = act(A @ W^T + bias) (+ addend).  A: (M,K) f16, W: (N,K) f16.
// Block = 256 threads = 8 waves; block tile 128x64; B tile staged in LDS
// (double-buffered) via async global->LDS copies; wave tile 16x64 (4 wmmas/step).
__global__ void wmma_gemm_nt(const _Float16* __restrict__ A, const _Float16* __restrict__ W,
                             const float* __restrict__ bias, const float* __restrict__ addend,
                             _Float16* __restrict__ out16, float* __restrict__ out32,
                             int M, int N, int K, int act) {
  __shared__ __align__(16) _Float16 bt[2][64 * 32];
  const int tid  = threadIdx.x;
  const int lane = tid & 31;
  const int wave = tid >> 5;
  const int ntiles = N >> 6;
  const int mtb = blockIdx.x / ntiles;
  const int nt  = blockIdx.x - mtb * ntiles;
  const int half = lane >> 4;
  const int l15  = lane & 15;

  // cooperative B-tile loader mapping: thread -> (row 0..63, 8-half chunk)
  const int lrow = tid >> 2;
  const int lcol = (tid & 3) * 8;
  const _Float16* gsrc = W + (size_t)(nt * 64 + lrow) * K + lcol;
  _Float16* ld0 = &bt[0][lrow * 32 + lcol];
  _Float16* ld1 = &bt[1][lrow * 32 + lcol];

  const int mrow = mtb * 128 + wave * 16;
  const _Float16* arow = A + (size_t)(mrow + l15) * K + half * 16;

  v8f acc[4] = {};
#if USE_ASYNC_LDS
  async_cp_b128(gsrc, ld0);
#endif
  for (int k0 = 0; k0 < K; k0 += 32) {
    const int ci = (k0 >> 5) & 1;
    _Float16* cbuf = &bt[ci][0];
#if USE_ASYNC_LDS
    wait_async0();          // this wave's portion of current tile landed
    __syncthreads();        // all portions landed; prev reads of other buffer done
    if (k0 + 32 < K) async_cp_b128(gsrc + k0 + 32, ci ? ld0 : ld1);
#else
    v8h tmp = *(const v8h*)(gsrc + k0);
    __syncthreads();        // prior reads of cbuf finished
    *(v8h*)(cbuf + lrow * 32 + lcol) = tmp;
    __syncthreads();        // tile visible
#endif
    v16h a = cat16(*(const v8h*)(arow + k0), *(const v8h*)(arow + k0 + 8));
#pragma unroll
    for (int s = 0; s < 4; ++s) {
      const _Float16* bp = cbuf + (s * 16 + l15) * 32 + half * 16;
      v16h bm = cat16(*(const v8h*)bp, *(const v8h*)(bp + 8));
      acc[s] = __builtin_amdgcn_wmma_f32_16x16x32_f16(false, a, false, bm,
                                                      (short)0, acc[s], false, false);
    }
  }
#pragma unroll
  for (int s = 0; s < 4; ++s) {
#pragma unroll
    for (int r = 0; r < 8; ++r) {
      int row = mrow + r + 8 * half;          // C/D layout: lanes 16-31 hold rows M+8
      int col = nt * 64 + s * 16 + l15;
      float v = acc[s][r];
      if (bias)   v += bias[col];
      if (act)    v = v * sigm(v);            // SiLU
      if (addend) v += addend[(size_t)row * N + col];
      if (out16)  out16[(size_t)row * N + col] = (_Float16)v;
      else        out32[(size_t)row * N + col] = v;
    }
  }
}

// ------------------------------------------------- gated GEMM: out = value * silu(gate)
// W: (2N,K); value = rows [0,N), gate = rows [N,2N). Same cooperative staging,
// tile holds 64 value rows + 64 gate rows (8KB per buffer).
__global__ void wmma_gemm_gated(const _Float16* __restrict__ A, const _Float16* __restrict__ W,
                                _Float16* __restrict__ out16, int M, int N, int K) {
  __shared__ __align__(16) _Float16 bt[2][128 * 32];
  const int tid  = threadIdx.x;
  const int lane = tid & 31;
  const int wave = tid >> 5;
  const int ntiles = N >> 6;
  const int mtb = blockIdx.x / ntiles;
  const int nt  = blockIdx.x - mtb * ntiles;
  const int half = lane >> 4;
  const int l15  = lane & 15;

  const int lrow = tid >> 2;
  const int lcol = (tid & 3) * 8;
  const _Float16* gsv = W + (size_t)(nt * 64 + lrow) * K + lcol;            // value rows
  const _Float16* gsg = W + (size_t)(N + nt * 64 + lrow) * K + lcol;        // gate rows
  _Float16* lv0 = &bt[0][lrow * 32 + lcol];
  _Float16* lg0 = &bt[0][(64 + lrow) * 32 + lcol];
  _Float16* lv1 = &bt[1][lrow * 32 + lcol];
  _Float16* lg1 = &bt[1][(64 + lrow) * 32 + lcol];

  const int mrow = mtb * 128 + wave * 16;
  const _Float16* arow = A + (size_t)(mrow + l15) * K + half * 16;

  v8f accv[4] = {}, accg[4] = {};
#if USE_ASYNC_LDS
  async_cp_b128(gsv, lv0);
  async_cp_b128(gsg, lg0);
#endif
  for (int k0 = 0; k0 < K; k0 += 32) {
    const int ci = (k0 >> 5) & 1;
    _Float16* cbuf = &bt[ci][0];
#if USE_ASYNC_LDS
    wait_async0();
    __syncthreads();
    if (k0 + 32 < K) {
      async_cp_b128(gsv + k0 + 32, ci ? lv0 : lv1);
      async_cp_b128(gsg + k0 + 32, ci ? lg0 : lg1);
    }
#else
    v8h tv = *(const v8h*)(gsv + k0);
    v8h tg = *(const v8h*)(gsg + k0);
    __syncthreads();
    *(v8h*)(cbuf + lrow * 32 + lcol)        = tv;
    *(v8h*)(cbuf + (64 + lrow) * 32 + lcol) = tg;
    __syncthreads();
#endif
    v16h a = cat16(*(const v8h*)(arow + k0), *(const v8h*)(arow + k0 + 8));
#pragma unroll
    for (int s = 0; s < 4; ++s) {
      const _Float16* bv = cbuf + (s * 16 + l15) * 32 + half * 16;
      const _Float16* bg = cbuf + (64 + s * 16 + l15) * 32 + half * 16;
      v16h bmv = cat16(*(const v8h*)bv, *(const v8h*)(bv + 8));
      v16h bmg = cat16(*(const v8h*)bg, *(const v8h*)(bg + 8));
      accv[s] = __builtin_amdgcn_wmma_f32_16x16x32_f16(false, a, false, bmv,
                                                       (short)0, accv[s], false, false);
      accg[s] = __builtin_amdgcn_wmma_f32_16x16x32_f16(false, a, false, bmg,
                                                       (short)0, accg[s], false, false);
    }
  }
#pragma unroll
  for (int s = 0; s < 4; ++s) {
#pragma unroll
    for (int r = 0; r < 8; ++r) {
      int row = mrow + r + 8 * half;
      int col = nt * 64 + s * 16 + l15;
      float v = accv[s][r];
      float g = accg[s][r];
      out16[(size_t)row * N + col] = (_Float16)(v * g * sigm(g));
    }
  }
}

// ------------------------------------------------- depthwise 3-tap conv over L (zero pad)
__global__ void dwconv3_k(const _Float16* __restrict__ Hin, const float* __restrict__ w,
                          const float* __restrict__ bias, _Float16* __restrict__ Hout) {
  int idx = blockIdx.x * blockDim.x + threadIdx.x;
  if (idx >= MTOK * DIMC) return;
  int c  = idx % DIMC;
  int bl = idx / DIMC;
  int l  = bl % LL;
  int b  = bl / LL;
  const _Float16* base = Hin + ((size_t)b * LL) * DIMC + c;
  float acc = bias[c];
  float w0 = w[3 * c + 0], w1 = w[3 * c + 1], w2 = w[3 * c + 2];
  if (l > 0)      acc += w0 * (float)base[(size_t)(l - 1) * DIMC];
  acc += w1 * (float)base[(size_t)l * DIMC];
  if (l < LL - 1) acc += w2 * (float)base[(size_t)(l + 1) * DIMC];
  Hout[idx] = (_Float16)acc;
}

// ------------------------------------------------- head split (+ l2norm / temperature / transpose)
// mode 0: l2norm, scale 1/temp -> (B,H,L,D)  (queries)
// mode 1: l2norm               -> (B,H,L,D)  (keys)
// mode 2: raw, transposed      -> (B,H,D,L)  (values)
__global__ void headify(const _Float16* __restrict__ P, _Float16* __restrict__ Out,
                        const float* __restrict__ temp, int mode) {
  int gw   = (blockIdx.x * blockDim.x + threadIdx.x) >> 5;   // one wave per (b,l,h)
  int lane = threadIdx.x & 31;
  if (gw >= BB * LL * NHEADS) return;
  int h = gw % NHEADS;
  int l = (gw / NHEADS) % LL;
  int b = gw / (NHEADS * LL);
  const _Float16* src = P + ((size_t)(b * LL + l)) * DIMC + h * HEADD;
  float x0 = (float)src[lane];
  float x1 = (float)src[lane + 32];
  float inv = 1.0f;
  if (mode < 2) {
    float ss = x0 * x0 + x1 * x1;
#pragma unroll
    for (int msk = 16; msk >= 1; msk >>= 1) ss += __shfl_xor(ss, msk, 32);
    float n = fmaxf(sqrtf(ss), 1e-12f);
    inv = 1.0f / n;
    if (mode == 0) inv *= 1.0f / temp[0];
  }
  _Float16 y0 = (_Float16)(x0 * inv);
  _Float16 y1 = (_Float16)(x1 * inv);
  if (mode == 2) {
    Out[((size_t)((b * NHEADS + h) * HEADD + lane)) * LL + l]      = y0;
    Out[((size_t)((b * NHEADS + h) * HEADD + lane + 32)) * LL + l] = y1;
  } else {
    _Float16* dst = Out + ((size_t)(b * NHEADS + h) * LL + l) * HEADD;
    dst[lane] = y0;
    dst[lane + 32] = y1;
  }
}

// ------------------------------------------------- fused flash attention (one wave = 16 q rows)
// Qn,Kn: (B,H,L,D) f16 (Q pre-scaled by 1/temp); Vt: (B,H,D,L) f16.
// Out = attn @ V  (head-merged (B,L,C) fp32) + residual X.
__global__ void attn_fused(const _Float16* __restrict__ Qn, const _Float16* __restrict__ Kn,
                           const _Float16* __restrict__ Vt, const float* __restrict__ Xres,
                           float* __restrict__ Out) {
  __shared__ __align__(16) _Float16 plds[8 * 512];   // per-wave 16x32 P staging
  const int lane = threadIdx.x & 31;
  const int wave = threadIdx.x >> 5;
  const int gw   = blockIdx.x * 8 + wave;
  const int qtiles = LL / 16;
  const int qt = gw % qtiles;
  const int h  = (gw / qtiles) % NHEADS;
  const int b  = gw / (qtiles * NHEADS);
  const int half = lane >> 4;
  const int l15  = lane & 15;
  const size_t bh = (size_t)b * NHEADS + h;

  const _Float16* qrow = Qn + (bh * LL + qt * 16 + l15) * HEADD;
  const v16h qa0 = cat16(*(const v8h*)(qrow + half * 16),
                         *(const v8h*)(qrow + 8 + half * 16));
  const v16h qa1 = cat16(*(const v8h*)(qrow + 32 + half * 16),
                         *(const v8h*)(qrow + 40 + half * 16));

  v8f o[4] = {};
  float mrun[8], lrun[8];
#pragma unroll
  for (int r = 0; r < 8; ++r) { mrun[r] = -1e30f; lrun[r] = 0.f; }

  _Float16* pb = plds + wave * 512;

  for (int kt = 0; kt < LL; kt += 32) {
    v8f s0 = {}, s1 = {};
    {
      const _Float16* kp = Kn + (bh * LL + kt + l15) * HEADD;
      v16h b0 = cat16(*(const v8h*)(kp + half * 16),      *(const v8h*)(kp + half * 16 + 8));
      v16h b1 = cat16(*(const v8h*)(kp + 32 + half * 16), *(const v8h*)(kp + 32 + half * 16 + 8));
      s0 = __builtin_amdgcn_wmma_f32_16x16x32_f16(false, qa0, false, b0, (short)0, s0, false, false);
      s0 = __builtin_amdgcn_wmma_f32_16x16x32_f16(false, qa1, false, b1, (short)0, s0, false, false);
    }
    {
      const _Float16* kp = Kn + (bh * LL + kt + 16 + l15) * HEADD;
      v16h b0 = cat16(*(const v8h*)(kp + half * 16),      *(const v8h*)(kp + half * 16 + 8));
      v16h b1 = cat16(*(const v8h*)(kp + 32 + half * 16), *(const v8h*)(kp + 32 + half * 16 + 8));
      s1 = __builtin_amdgcn_wmma_f32_16x16x32_f16(false, qa0, false, b0, (short)0, s1, false, false);
      s1 = __builtin_amdgcn_wmma_f32_16x16x32_f16(false, qa1, false, b1, (short)0, s1, false, false);
    }
    float p0[8], p1[8];
#pragma unroll
    for (int r = 0; r < 8; ++r) {
      float mx = fmaxf(s0[r], s1[r]);
#pragma unroll
      for (int msk = 8; msk >= 1; msk >>= 1) mx = fmaxf(mx, __shfl_xor(mx, msk, 32));
      float mnew = fmaxf(mrun[r], mx);
      float sc = __expf(mrun[r] - mnew);
      p0[r] = __expf(s0[r] - mnew);
      p1[r] = __expf(s1[r] - mnew);
      float rs = p0[r] + p1[r];
#pragma unroll
      for (int msk = 8; msk >= 1; msk >>= 1) rs += __shfl_xor(rs, msk, 32);
      lrun[r] = lrun[r] * sc + rs;
      mrun[r] = mnew;
#pragma unroll
      for (int t = 0; t < 4; ++t) o[t][r] *= sc;
    }
#pragma unroll
    for (int r = 0; r < 8; ++r) {
      int row = r + 8 * half;
      pb[row * 32 + l15]      = (_Float16)p0[r];
      pb[row * 32 + 16 + l15] = (_Float16)p1[r];
    }
    v16h pa = cat16(*(const v8h*)(pb + l15 * 32 + half * 16),
                    *(const v8h*)(pb + l15 * 32 + 8 + half * 16));
#pragma unroll
    for (int t = 0; t < 4; ++t) {
      const _Float16* vp = Vt + (bh * HEADD + t * 16 + l15) * LL + kt + half * 16;
      v16h vb = cat16(*(const v8h*)vp, *(const v8h*)(vp + 8));
      o[t] = __builtin_amdgcn_wmma_f32_16x16x32_f16(false, pa, false, vb, (short)0, o[t], false, false);
    }
  }

#pragma unroll
  for (int r = 0; r < 8; ++r) lrun[r] = 1.0f / lrun[r];
#pragma unroll
  for (int t = 0; t < 4; ++t)
#pragma unroll
    for (int r = 0; r < 8; ++r) {
      int tok = qt * 16 + r + 8 * half;
      int col = h * HEADD + t * 16 + l15;
      size_t idx = ((size_t)b * LL + tok) * DIMC + col;
      Out[idx] = o[t][r] * lrun[r] + Xres[idx];
    }
}

// ------------------------------------------------- layernorm over C=1024 (one block per row)
__global__ void layernorm_k(const float* __restrict__ X, const float* __restrict__ g,
                            const float* __restrict__ bb, _Float16* __restrict__ out16,
                            float* __restrict__ out32) {
  __shared__ float sh[16];
  const int row = blockIdx.x;
  const float* x = X + (size_t)row * DIMC;
  float s = 0.f, s2 = 0.f;
  for (int i = threadIdx.x; i < DIMC; i += blockDim.x) {
    float v = x[i]; s += v; s2 += v * v;
  }
#pragma unroll
  for (int msk = 16; msk >= 1; msk >>= 1) {
    s  += __shfl_xor(s,  msk, 32);
    s2 += __shfl_xor(s2, msk, 32);
  }
  int wave = threadIdx.x >> 5, lane = threadIdx.x & 31;
  if (lane == 0) { sh[wave] = s; sh[8 + wave] = s2; }
  __syncthreads();
  float ts = 0.f, ts2 = 0.f;
#pragma unroll
  for (int w = 0; w < 8; ++w) { ts += sh[w]; ts2 += sh[8 + w]; }
  float mean = ts / DIMC;
  float var  = ts2 / DIMC - mean * mean;
  float inv  = rsqrtf(var + LN_EPS);
  for (int i = threadIdx.x; i < DIMC; i += blockDim.x) {
    float y = (x[i] - mean) * inv * g[i] + bb[i];
    if (out16) out16[(size_t)row * DIMC + i] = (_Float16)y;
    if (out32) out32[(size_t)row * DIMC + i] = y;
  }
}

// ================================================================ host side
extern "C" void kernel_launch(void* const* d_in, const int* in_sizes, int n_in,
                              void* d_out, int out_size, void* d_ws, size_t ws_size,
                              hipStream_t stream) {
  const float* x      = (const float*)d_in[0];
  const float* wq     = (const float*)d_in[1];
  const float* wk     = (const float*)d_in[2];
  const float* wv     = (const float*)d_in[3];
  const float* q_dw_w = (const float*)d_in[4];
  const float* q_dw_b = (const float*)d_in[5];
  const float* q_pw_w = (const float*)d_in[6];
  const float* q_pw_b = (const float*)d_in[7];
  const float* k_dw_w = (const float*)d_in[8];
  const float* k_dw_b = (const float*)d_in[9];
  const float* k_pw_w = (const float*)d_in[10];
  const float* k_pw_b = (const float*)d_in[11];
  const float* v_dw_w = (const float*)d_in[12];
  const float* v_dw_b = (const float*)d_in[13];
  const float* v_pw_w = (const float*)d_in[14];
  const float* v_pw_b = (const float*)d_in[15];
  const float* gn_g   = (const float*)d_in[16];
  const float* gn_b   = (const float*)d_in[17];
  const float* w_gate = (const float*)d_in[18];
  const float* w_out  = (const float*)d_in[19];
  const float* b_out  = (const float*)d_in[20];
  const float* temp   = (const float*)d_in[21];
  const float* fn_g   = (const float*)d_in[22];
  const float* fn_b   = (const float*)d_in[23];
  float* out = (float*)d_out;

  char* ws = (char*)d_ws;
  size_t off = 0;
  auto Aw = [&](size_t bytes) -> char* {
    char* p = ws + off;
    off += (bytes + 255) & ~(size_t)255;
    return p;
  };
  const size_t E  = (size_t)MTOK * DIMC;
  _Float16* x16   = (_Float16*)Aw(E * 2);
  _Float16* wq16  = (_Float16*)Aw((size_t)DIMC * DIMC * 2);
  _Float16* wk16  = (_Float16*)Aw((size_t)DIMC * DIMC * 2);
  _Float16* wv16  = (_Float16*)Aw((size_t)DIMC * DIMC * 2);
  _Float16* qpw16 = (_Float16*)Aw((size_t)DIMC * DIMC * 2);
  _Float16* kpw16 = (_Float16*)Aw((size_t)DIMC * DIMC * 2);
  _Float16* vpw16 = (_Float16*)Aw((size_t)DIMC * DIMC * 2);
  _Float16* wg16  = (_Float16*)Aw((size_t)2 * DIMC * DIMC * 2);
  _Float16* wo16  = (_Float16*)Aw((size_t)DIMC * DIMC * 2);
  _Float16* h16   = (_Float16*)Aw(E * 2);
  _Float16* dcv16 = (_Float16*)Aw(E * 2);
  _Float16* p16   = (_Float16*)Aw(E * 2);
  _Float16* Qn    = (_Float16*)Aw(E * 2);
  _Float16* Kn    = (_Float16*)Aw(E * 2);
  _Float16* Vt    = (_Float16*)Aw(E * 2);
  float*    attn  = (float*)Aw(E * 4);
  _Float16* ln1   = (_Float16*)Aw(E * 2);
  _Float16* gated = (_Float16*)Aw(E * 2);
  float*    y2    = (float*)Aw(E * 4);

  auto cvt = [&](const float* in, _Float16* o, size_t n) {
    cvt_f32_f16<<<(int)((n + 255) / 256), 256, 0, stream>>>(in, o, (int)n);
  };
  cvt(x, x16, E);
  cvt(wq, wq16, (size_t)DIMC * DIMC);
  cvt(wk, wk16, (size_t)DIMC * DIMC);
  cvt(wv, wv16, (size_t)DIMC * DIMC);
  cvt(q_pw_w, qpw16, (size_t)DIMC * DIMC);
  cvt(k_pw_w, kpw16, (size_t)DIMC * DIMC);
  cvt(v_pw_w, vpw16, (size_t)DIMC * DIMC);
  cvt(w_gate, wg16, (size_t)2 * DIMC * DIMC);
  cvt(w_out, wo16, (size_t)DIMC * DIMC);

  const int gemm_blocks = (MTOK / 128) * (DIMC / 64);      // 512
  const int dw_blocks   = (MTOK * DIMC + 255) / 256;
  const int head_blocks = (BB * LL * NHEADS * 32) / 256;   // 8192

  // ---- Q chain
  wmma_gemm_nt<<<gemm_blocks, 256, 0, stream>>>(x16, wq16, nullptr, nullptr, h16, nullptr,
                                                MTOK, DIMC, DIMC, 1);
  dwconv3_k<<<dw_blocks, 256, 0, stream>>>(h16, q_dw_w, q_dw_b, dcv16);
  wmma_gemm_nt<<<gemm_blocks, 256, 0, stream>>>(dcv16, qpw16, q_pw_b, nullptr, p16, nullptr,
                                                MTOK, DIMC, DIMC, 0);
  headify<<<head_blocks, 256, 0, stream>>>(p16, Qn, temp, 0);
  // ---- K chain
  wmma_gemm_nt<<<gemm_blocks, 256, 0, stream>>>(x16, wk16, nullptr, nullptr, h16, nullptr,
                                                MTOK, DIMC, DIMC, 1);
  dwconv3_k<<<dw_blocks, 256, 0, stream>>>(h16, k_dw_w, k_dw_b, dcv16);
  wmma_gemm_nt<<<gemm_blocks, 256, 0, stream>>>(dcv16, kpw16, k_pw_b, nullptr, p16, nullptr,
                                                MTOK, DIMC, DIMC, 0);
  headify<<<head_blocks, 256, 0, stream>>>(p16, Kn, temp, 1);
  // ---- V chain
  wmma_gemm_nt<<<gemm_blocks, 256, 0, stream>>>(x16, wv16, nullptr, nullptr, h16, nullptr,
                                                MTOK, DIMC, DIMC, 1);
  dwconv3_k<<<dw_blocks, 256, 0, stream>>>(h16, v_dw_w, v_dw_b, dcv16);
  wmma_gemm_nt<<<gemm_blocks, 256, 0, stream>>>(dcv16, vpw16, v_pw_b, nullptr, p16, nullptr,
                                                MTOK, DIMC, DIMC, 0);
  headify<<<head_blocks, 256, 0, stream>>>(p16, Vt, temp, 2);

  // ---- attention (+ residual)
  const int attn_blocks = (BB * NHEADS * (LL / 16)) / 8;   // 512
  attn_fused<<<attn_blocks, 256, 0, stream>>>(Qn, Kn, Vt, x, attn);

  // ---- layernorm 1 -> f16
  layernorm_k<<<MTOK, 256, 0, stream>>>(attn, gn_g, gn_b, ln1, nullptr);

  // ---- gate GEMM: value * silu(gate)
  wmma_gemm_gated<<<gemm_blocks, 256, 0, stream>>>(ln1, wg16, gated, MTOK, DIMC, DIMC);

  // ---- out GEMM (+ b_out + attn residual) -> fp32
  wmma_gemm_nt<<<gemm_blocks, 256, 0, stream>>>(gated, wo16, b_out, attn, nullptr, y2,
                                                MTOK, DIMC, DIMC, 0);

  // ---- final layernorm -> d_out fp32
  layernorm_k<<<MTOK, 256, 0, stream>>>(y2, fn_g, fn_b, nullptr, out);
}
